// GNNMOT_71975061946762
// MI455X (gfx1250) — matrix operand order repaired and checked by reading, not compiled, not run on hardware
//
#include <hip/hip_runtime.h>
#include <math.h>

typedef __attribute__((ext_vector_type(16))) _Float16 v16h;
typedef __attribute__((ext_vector_type(8)))  float    v8f;

#define NDET 256
#define NTRK 256
#define NPTS 1024
#define TSEQ 10
#define VTOT 512

// ---- WMMA fragment index helpers (CDNA5 16x16x32 f16 layouts, ISA 7.12.2) ----
__device__ __forceinline__ int ak_idx(int lane, int e) {
    int hi = lane >> 4;
    return (e < 8) ? (8 * hi + e) : (16 + 8 * hi + (e - 8));
}
__device__ __forceinline__ int bk_idx(int lane, int e) {
    int hi = lane >> 4;
    return (e < 8) ? (8 * hi + e) : (16 + 8 * hi + (e - 8));   // same k order as A
}
__device__ __forceinline__ v8f wmma16(v16h a, v16h b, v8f c) {
    return __builtin_amdgcn_wmma_f32_16x16x32_f16(false, a, false, b, (short)0, c, false, false);
}

// =====================================================================
// PointNet: 512 boxes x 1024 pts x 5 -> 64 -> 128 -> 64, maxpool over pts.
// One block per box, 8 waves, each wave handles 16-point tiles via WMMA.
// Writes appearance feature into g[box][0:64].
// =====================================================================
__global__ void __launch_bounds__(256) pn_kernel(
    const float* __restrict__ det_pc, const float* __restrict__ trk_pc,
    const float* __restrict__ W1, const float* __restrict__ B1,
    const float* __restrict__ W2, const float* __restrict__ B2,
    const float* __restrict__ W3, const float* __restrict__ B3,
    float* __restrict__ g)
{
    __shared__ _Float16 w1h[64 * 5];
    __shared__ _Float16 w2h[128 * 64];
    __shared__ _Float16 w3h[64 * 128];
    __shared__ float b1s[64], b2s[128], b3s[64];
    __shared__ _Float16 stage[8 * 16 * 128];   // per-wave 16x128 staging
    __shared__ float red[8 * 64];

    const int tid = threadIdx.x;
    const int box = blockIdx.x;
    const float* pc = (box < NDET) ? (det_pc + (size_t)box * NPTS * 5)
                                   : (trk_pc + (size_t)(box - NDET) * NPTS * 5);

    for (int i = tid; i < 64 * 5;   i += 256) w1h[i] = (_Float16)W1[i];
    for (int i = tid; i < 128 * 64; i += 256) w2h[i] = (_Float16)W2[i];
    for (int i = tid; i < 64 * 128; i += 256) w3h[i] = (_Float16)W3[i];
    if (tid < 64)  b1s[tid] = B1[tid];
    if (tid < 128) b2s[tid] = B2[tid];
    if (tid < 64)  b3s[tid] = B3[tid];
    __syncthreads();

    const int wave = tid >> 5, lane = tid & 31;
    const int m = lane & 15, hi = lane >> 4, n = lane & 15;
    _Float16* st = stage + wave * 16 * 128;

    float runmax[4][8];
#pragma unroll
    for (int j = 0; j < 4; ++j)
#pragma unroll
        for (int r = 0; r < 8; ++r) runmax[j][r] = -3.4e38f;

    for (int t = wave; t < NPTS / 16; t += 8) {
        // ---- layer 1: 5 -> 64 (K zero-padded to 32) ----
        v16h a;
#pragma unroll
        for (int e = 0; e < 16; ++e) {
            int k = ak_idx(lane, e);
            a[e] = (k < 5) ? (_Float16)pc[(size_t)(t * 16 + m) * 5 + k] : (_Float16)0.0f;
        }
#pragma unroll
        for (int j = 0; j < 4; ++j) {
            v16h b;
#pragma unroll
            for (int e = 0; e < 16; ++e) {
                int k = bk_idx(lane, e);
                b[e] = (k < 5) ? w1h[(j * 16 + n) * 5 + k] : (_Float16)0.0f;
            }
            v8f c;
            float bias = b1s[j * 16 + n];
#pragma unroll
            for (int r = 0; r < 8; ++r) c[r] = bias;
            c = wmma16(a, b, c);
#pragma unroll
            for (int r = 0; r < 8; ++r) {
                float v = c[r]; v = v > 0.f ? v : 0.f;
                st[(r + 8 * hi) * 128 + (j * 16 + n)] = (_Float16)v;
            }
        }
        // ---- layer 2: 64 -> 128 ----
        v16h a0, a1;
#pragma unroll
        for (int e = 0; e < 16; ++e) {
            a0[e] = st[m * 128 + ak_idx(lane, e)];
            a1[e] = st[m * 128 + 32 + ak_idx(lane, e)];
        }
#pragma unroll
        for (int j = 0; j < 8; ++j) {
            v8f c;
            float bias = b2s[j * 16 + n];
#pragma unroll
            for (int r = 0; r < 8; ++r) c[r] = bias;
            v16h b;
#pragma unroll
            for (int e = 0; e < 16; ++e) b[e] = w2h[(j * 16 + n) * 64 + bk_idx(lane, e)];
            c = wmma16(a0, b, c);
#pragma unroll
            for (int e = 0; e < 16; ++e) b[e] = w2h[(j * 16 + n) * 64 + 32 + bk_idx(lane, e)];
            c = wmma16(a1, b, c);
#pragma unroll
            for (int r = 0; r < 8; ++r) {
                float v = c[r]; v = v > 0.f ? v : 0.f;
                st[(r + 8 * hi) * 128 + (j * 16 + n)] = (_Float16)v;
            }
        }
        // ---- layer 3: 128 -> 64 (no relu) + running max-pool ----
        v16h A[4];
#pragma unroll
        for (int kk = 0; kk < 4; ++kk)
#pragma unroll
            for (int e = 0; e < 16; ++e)
                A[kk][e] = st[m * 128 + kk * 32 + ak_idx(lane, e)];
#pragma unroll
        for (int j = 0; j < 4; ++j) {
            v8f c;
            float bias = b3s[j * 16 + n];
#pragma unroll
            for (int r = 0; r < 8; ++r) c[r] = bias;
#pragma unroll
            for (int kk = 0; kk < 4; ++kk) {
                v16h b;
#pragma unroll
                for (int e = 0; e < 16; ++e)
                    b[e] = w3h[(j * 16 + n) * 128 + kk * 32 + bk_idx(lane, e)];
                c = wmma16(A[kk], b, c);
            }
#pragma unroll
            for (int r = 0; r < 8; ++r) runmax[j][r] = fmaxf(runmax[j][r], c[r]);
        }
    }

    // reduce max over points: over r, over lane-halves, then over waves
#pragma unroll
    for (int j = 0; j < 4; ++j) {
        float mx = runmax[j][0];
#pragma unroll
        for (int r = 1; r < 8; ++r) mx = fmaxf(mx, runmax[j][r]);
        mx = fmaxf(mx, __shfl_xor(mx, 16));
        if (lane < 16) red[wave * 64 + j * 16 + lane] = mx;
    }
    __syncthreads();
    if (tid < 64) {
        float mx = red[tid];
#pragma unroll
        for (int w = 1; w < 8; ++w) mx = fmaxf(mx, red[w * 64 + tid]);
        g[(size_t)box * 128 + tid] = mx;
    }
}

// =====================================================================
// Det motion MLP 9->32->64, writes g[d][64:128]
// =====================================================================
__global__ void __launch_bounds__(256) dm_kernel(
    const float* __restrict__ boxes,
    const float* __restrict__ W1, const float* __restrict__ B1,
    const float* __restrict__ W2, const float* __restrict__ B2,
    float* __restrict__ g)
{
    int d = threadIdx.x;
    float x[9];
#pragma unroll
    for (int k = 0; k < 9; ++k) x[k] = boxes[d * 9 + k];
    float h1[32];
#pragma unroll
    for (int o = 0; o < 32; ++o) {
        float s = B1[o];
#pragma unroll
        for (int k = 0; k < 9; ++k) s += W1[o * 9 + k] * x[k];
        h1[o] = s > 0.f ? s : 0.f;
    }
    for (int o = 0; o < 64; ++o) {
        float s = B2[o];
#pragma unroll
        for (int k = 0; k < 32; ++k) s += W2[o * 32 + k] * h1[k];
        g[(size_t)d * 128 + 64 + o] = s;
    }
}

// =====================================================================
// 2-layer LSTM, hidden 64, torch gate order (i,f,g,o).
// One block per track, 64 threads = hidden units. Writes g[256+trk][64:128].
// =====================================================================
__global__ void __launch_bounds__(64) lstm_kernel(
    const float* __restrict__ boxes,
    const float* __restrict__ Wih0, const float* __restrict__ Whh0,
    const float* __restrict__ bih0, const float* __restrict__ bhh0,
    const float* __restrict__ Wih1, const float* __restrict__ Whh1,
    const float* __restrict__ bih1, const float* __restrict__ bhh1,
    float* __restrict__ g)
{
    __shared__ float hbuf[64];
    __shared__ float ys[TSEQ * 64];
    const int u = threadIdx.x;
    const int trk = blockIdx.x;

    // layer 0
    float c = 0.f;
    hbuf[u] = 0.f;
    __syncthreads();
    for (int t = 0; t < TSEQ; ++t) {
        const float* x = boxes + (size_t)trk * TSEQ * 9 + t * 9;
        float z[4];
#pragma unroll
        for (int gg = 0; gg < 4; ++gg) {
            int row = gg * 64 + u;
            float s = bih0[row] + bhh0[row];
#pragma unroll
            for (int k = 0; k < 9; ++k) s += Wih0[row * 9 + k] * x[k];
            for (int k = 0; k < 64; ++k) s += Whh0[row * 64 + k] * hbuf[k];
            z[gg] = s;
        }
        float ig = 1.f / (1.f + expf(-z[0]));
        float fg = 1.f / (1.f + expf(-z[1]));
        float gt = tanhf(z[2]);
        float og = 1.f / (1.f + expf(-z[3]));
        c = fg * c + ig * gt;
        float hn = og * tanhf(c);
        __syncthreads();
        hbuf[u] = hn; ys[t * 64 + u] = hn;
        __syncthreads();
    }
    // layer 1
    c = 0.f;
    float hlast = 0.f;
    __syncthreads();
    hbuf[u] = 0.f;
    __syncthreads();
    for (int t = 0; t < TSEQ; ++t) {
        const float* x = ys + t * 64;
        float z[4];
#pragma unroll
        for (int gg = 0; gg < 4; ++gg) {
            int row = gg * 64 + u;
            float s = bih1[row] + bhh1[row];
            for (int k = 0; k < 64; ++k) s += Wih1[row * 64 + k] * x[k];
            for (int k = 0; k < 64; ++k) s += Whh1[row * 64 + k] * hbuf[k];
            z[gg] = s;
        }
        float ig = 1.f / (1.f + expf(-z[0]));
        float fg = 1.f / (1.f + expf(-z[1]));
        float gt = tanhf(z[2]);
        float og = 1.f / (1.f + expf(-z[3]));
        c = fg * c + ig * gt;
        hlast = og * tanhf(c);
        __syncthreads();
        hbuf[u] = hlast;
        __syncthreads();
    }
    g[(size_t)(NDET + trk) * 128 + 64 + u] = hlast;
}

// =====================================================================
// EdgeConv GEMMs: t = h@Wt^T, u = h@Wp^T + bp + bt - t  (512x128, K=128)
// 1 wave per 16x16 tile. grid 64 x 128 threads.
// =====================================================================
__global__ void __launch_bounds__(128) ec_gemm_kernel(
    const float* __restrict__ h,
    const float* __restrict__ Wt, const float* __restrict__ bt,
    const float* __restrict__ Wp, const float* __restrict__ bp,
    float* __restrict__ tbuf, float* __restrict__ ubuf)
{
    const int lane = threadIdx.x & 31;
    const int wid = blockIdx.x * 4 + (threadIdx.x >> 5);   // 0..255
    const int mt = wid >> 3, nt = wid & 7;
    const int m = lane & 15, hi = lane >> 4, n = lane & 15;

    v16h A[4];
#pragma unroll
    for (int kk = 0; kk < 4; ++kk)
#pragma unroll
        for (int e = 0; e < 16; ++e)
            A[kk][e] = (_Float16)h[(size_t)(mt * 16 + m) * 128 + kk * 32 + ak_idx(lane, e)];

    v8f ct, cp;
#pragma unroll
    for (int r = 0; r < 8; ++r) { ct[r] = 0.f; cp[r] = 0.f; }
#pragma unroll
    for (int kk = 0; kk < 4; ++kk) {
        v16h b;
#pragma unroll
        for (int e = 0; e < 16; ++e)
            b[e] = (_Float16)Wt[(size_t)(nt * 16 + n) * 128 + kk * 32 + bk_idx(lane, e)];
        ct = wmma16(A[kk], b, ct);
    }
#pragma unroll
    for (int kk = 0; kk < 4; ++kk) {
        v16h b;
#pragma unroll
        for (int e = 0; e < 16; ++e)
            b[e] = (_Float16)Wp[(size_t)(nt * 16 + n) * 128 + kk * 32 + bk_idx(lane, e)];
        cp = wmma16(A[kk], b, cp);
    }
    const int col = nt * 16 + n;
    const float bb = bp[col] + bt[col];
#pragma unroll
    for (int r = 0; r < 8; ++r) {
        int row = mt * 16 + r + 8 * hi;
        tbuf[(size_t)row * 128 + col] = ct[r];
        ubuf[(size_t)row * 128 + col] = cp[r] + bb - ct[r];
    }
}

// =====================================================================
// EdgeConv masked max + relu: h'[i][c] = relu(u[i][c] + max_{mask} t[j][c])
// mask_in[i][j] = adj[j][i]!=0 || i==j.  grid 512 x 128.
// =====================================================================
__global__ void __launch_bounds__(128) ec_maxrelu_kernel(
    const float* __restrict__ tbuf, const float* __restrict__ ubuf,
    const int* __restrict__ adj, float* __restrict__ hout)
{
    __shared__ unsigned char msk[VTOT];
    const int i = blockIdx.x, c = threadIdx.x;
    for (int j = c; j < VTOT; j += 128)
        msk[j] = (adj[(size_t)j * VTOT + i] != 0) || (j == i);
    __syncthreads();
    float mx = -3.4e38f;
    for (int j = 0; j < VTOT; ++j)
        if (msk[j]) mx = fmaxf(mx, tbuf[(size_t)j * 128 + c]);
    float v = ubuf[(size_t)i * 128 + c] + mx;
    hout[(size_t)i * 128 + c] = v > 0.f ? v : 0.f;
}

// =====================================================================
// Edge regression: vals[i][j] = W2·relu(W1(h[j]-h[i])+b1)+b2 ; aff with mask.
// 1 wave per (i, 16-j tile): A built as diff on the fly, K=128, N=64,
// then the 64->1 layer folded in via cross-lane reduction.
// grid 1024 x 128 threads.
// =====================================================================
__global__ void __launch_bounds__(128) er_kernel(
    const float* __restrict__ h, const int* __restrict__ adj,
    const float* __restrict__ W1, const float* __restrict__ B1,
    const float* __restrict__ W2, const float* __restrict__ B2,
    float* __restrict__ aff)
{
    const int lane = threadIdx.x & 31;
    const int wid = blockIdx.x * 4 + (threadIdx.x >> 5);   // 0..4095
    const int i = wid >> 4, jt = wid & 15;
    const int m = lane & 15, hi = lane >> 4, n = lane & 15;

    v16h A[4];
#pragma unroll
    for (int kk = 0; kk < 4; ++kk)
#pragma unroll
        for (int e = 0; e < 16; ++e) {
            int k = kk * 32 + ak_idx(lane, e);
            A[kk][e] = (_Float16)(h[(size_t)(jt * 16 + m) * 128 + k] - h[(size_t)i * 128 + k]);
        }

    float part[8];
#pragma unroll
    for (int r = 0; r < 8; ++r) part[r] = 0.f;

#pragma unroll
    for (int ntl = 0; ntl < 4; ++ntl) {
        v8f c;
        float bias = B1[ntl * 16 + n];
#pragma unroll
        for (int r = 0; r < 8; ++r) c[r] = bias;
#pragma unroll
        for (int kk = 0; kk < 4; ++kk) {
            v16h b;
#pragma unroll
            for (int e = 0; e < 16; ++e)
                b[e] = (_Float16)W1[(size_t)(ntl * 16 + n) * 128 + kk * 32 + bk_idx(lane, e)];
            c = wmma16(A[kk], b, c);
        }
        float w2 = W2[ntl * 16 + n];
#pragma unroll
        for (int r = 0; r < 8; ++r) {
            float v = c[r]; v = v > 0.f ? v : 0.f;
            part[r] += v * w2;
        }
    }
    // reduce over the 16 lanes of each half (N dimension)
#pragma unroll
    for (int off = 1; off < 16; off <<= 1)
#pragma unroll
        for (int r = 0; r < 8; ++r) part[r] += __shfl_xor(part[r], off);

    const float b2 = B2[0];
    if (n == 0) {
#pragma unroll
        for (int r = 0; r < 8; ++r) {
            int j = jt * 16 + r + 8 * hi;
            float val = part[r] + b2;
            bool em = adj[(size_t)i * VTOT + (NDET + j)] != 0;
            aff[(size_t)i * 256 + j] = em ? val : 10.0f;
        }
    }
}

// =====================================================================
// Greedy matching: sequential over rows, argmin (first-min tiebreak).
// =====================================================================
__global__ void __launch_bounds__(256) greedy_kernel(
    const float* __restrict__ aff, float* __restrict__ out)
{
    __shared__ float sv[256];
    __shared__ int si[256];
    __shared__ unsigned char taken[256];
    const int tid = threadIdx.x;
    taken[tid] = 0;
    __syncthreads();
    for (int i = 0; i < NDET; ++i) {
        float v = taken[tid] ? 10.0f : aff[(size_t)i * 256 + tid];
        sv[tid] = v; si[tid] = tid;
        __syncthreads();
        for (int s = 128; s > 0; s >>= 1) {
            if (tid < s) {
                float v2 = sv[tid + s]; int i2 = si[tid + s];
                if (v2 < sv[tid] || (v2 == sv[tid] && i2 < si[tid])) { sv[tid] = v2; si[tid] = i2; }
            }
            __syncthreads();
        }
        if (tid == 0) {
            bool ok = sv[0] < 10.0f;
            if (ok) taken[si[0]] = 1;
            out[i] = ok ? (float)si[0] : -1.0f;
        }
        __syncthreads();
    }
}

// =====================================================================
extern "C" void kernel_launch(void* const* d_in, const int* in_sizes, int n_in,
                              void* d_out, int out_size, void* d_ws, size_t ws_size,
                              hipStream_t stream)
{
    (void)in_sizes; (void)n_in; (void)out_size; (void)ws_size;

    const float* det_pc  = (const float*)d_in[0];
    const float* det_box = (const float*)d_in[1];
    const float* trk_pc  = (const float*)d_in[2];
    const float* trk_box = (const float*)d_in[3];
    const int*   adj     = (const int*)  d_in[4];

    const float* P[38];
    for (int i = 0; i < 38; ++i) P[i] = (const float*)d_in[5 + i];
    // params in make_params insertion order
    const float* pn_W1 = P[0];  const float* pn_b1 = P[1];
    const float* pn_W2 = P[2];  const float* pn_b2 = P[3];
    const float* pn_W3 = P[4];  const float* pn_b3 = P[5];
    const float* dm_W1 = P[6];  const float* dm_b1 = P[7];
    const float* dm_W2 = P[8];  const float* dm_b2 = P[9];
    const float* l0_Wih = P[10]; const float* l0_Whh = P[11];
    const float* l0_bih = P[12]; const float* l0_bhh = P[13];
    const float* l1_Wih = P[14]; const float* l1_Whh = P[15];
    const float* l1_bih = P[16]; const float* l1_bhh = P[17];
    const float* ecW[4][4];
    for (int l = 0; l < 4; ++l)
        for (int q = 0; q < 4; ++q) ecW[l][q] = P[18 + l * 4 + q];  // Wt, bt, Wp, bp
    const float* er_W1 = P[34]; const float* er_b1 = P[35];
    const float* er_W2 = P[36]; const float* er_b2 = P[37];

    float* ws   = (float*)d_ws;
    float* g    = ws;                 // 512*128 graph features (= h ping buffer A)
    float* hB   = ws + 65536;         // h ping buffer B
    float* tbuf = ws + 131072;        // 512*128
    float* ubuf = ws + 196608;        // 512*128
    float* aff     = (float*)d_out;   // 256*256
    float* matched = aff + 65536;     // 256

    pn_kernel<<<VTOT, 256, 0, stream>>>(det_pc, trk_pc, pn_W1, pn_b1, pn_W2, pn_b2, pn_W3, pn_b3, g);
    dm_kernel<<<1, 256, 0, stream>>>(det_box, dm_W1, dm_b1, dm_W2, dm_b2, g);
    lstm_kernel<<<NTRK, 64, 0, stream>>>(trk_box, l0_Wih, l0_Whh, l0_bih, l0_bhh,
                                         l1_Wih, l1_Whh, l1_bih, l1_bhh, g);
    float* hin = g; float* hout = hB;
    for (int l = 0; l < 4; ++l) {
        ec_gemm_kernel<<<64, 128, 0, stream>>>(hin, ecW[l][0], ecW[l][1], ecW[l][2], ecW[l][3],
                                               tbuf, ubuf);
        ec_maxrelu_kernel<<<VTOT, 128, 0, stream>>>(tbuf, ubuf, adj, hout);
        float* tmp = hin; hin = hout; hout = tmp;
    }
    er_kernel<<<1024, 128, 0, stream>>>(hin, adj, er_W1, er_b1, er_W2, er_b2, aff);
    greedy_kernel<<<1, 256, 0, stream>>>(aff, matched);
}
